// NeighConsensus_32031866094248
// MI455X (gfx1250) — matrix operand order repaired and checked by reading, not compiled, not run on hardware
//
#include <hip/hip_runtime.h>

typedef _Float16 half_t;
typedef __attribute__((ext_vector_type(16))) _Float16 v16h;
typedef __attribute__((ext_vector_type(8)))  _Float16 v8h;
typedef __attribute__((ext_vector_type(8)))  float    v8f;

#define SDIM 30
#define PTOT 1620000            // 2 * 30^4 positions
#define CH 16                   // channels padded to 16
#define TAPS 81                 // 3^4
#define CHUNKS 41               // 82 taps / 2 taps-per-chunk (K=32 = 2 taps * 16 ch)
#define NTILES (PTOT / 16)      // 101250 (exact)
#define JTILES 4                // tiles per wave (register-pressure sweet spot)
#define NWAVES ((NTILES + JTILES - 1) / JTILES)   // 25313 (last wave clamps)
#define WBUF_HALVES (CHUNKS * 32 * 16)   // 20992 halves per packed-weight buffer

// ---- ws layout (bytes) ------------------------------------------------------
#define WS_WPACK   0u
#define WS_BIAS    251904u                 // 6*20992*2
#define WS_OFFTAB  252096u                 // 48*4 after bias
#define WS_ACT0    252672u                 // 82*4=328 after offtab, 256-aligned
#define ACT_BYTES  51840000u               // PTOT*CH*2
#define WS_ACT1    (WS_ACT0 + ACT_BYTES)
#define WS_ACT2    (WS_ACT1 + ACT_BYTES)
#define WS_ZPAGE   (WS_ACT2 + ACT_BYTES)   // 32 B of zeros

// ---------------- input packing: x fp32 [P] -> act0 f16 [P][16] (ch0 = x) ----
__global__ void pack_input_kernel(const float* __restrict__ x, half_t* __restrict__ act0) {
  int i = blockIdx.x * blockDim.x + threadIdx.x;
  if (i >= PTOT) return;
  v8h lo = {};
  lo[0] = (half_t)x[i];
  v8h hi = {};
  *(v8h*)(act0 + (size_t)i * CH) = lo;
  *(v8h*)(act0 + (size_t)i * CH + 8) = hi;
}

// ---------------- weight packing into WMMA A-fragment order ------------------
// 6 buffers: q = branch*3 + layer.  Buffer layout: [chunk][lane][e0..15] halves.
// A element (lane,e): M = lane&15, kloc = 16*(lane>>4)+e.
// K-group gk = kloc>>4 selects tap = chunk + 41*gk; cin = kloc&15.
__global__ void pack_weights_kernel(const float* __restrict__ w1,
                                    const float* __restrict__ w2,
                                    const float* __restrict__ w3,
                                    half_t* __restrict__ wpack) {
  int tid = blockIdx.x * blockDim.x + threadIdx.x;
  if (tid >= 6 * WBUF_HALVES) return;
  int q = tid / WBUF_HALVES;
  int r = tid % WBUF_HALVES;
  int lane = (r >> 4) & 31;
  int e = r & 15;
  int chunk = r >> 9;
  int layer = q % 3, perm = q / 3;
  const float* w = (layer == 0) ? w1 : (layer == 1) ? w2 : w3;
  int Cin  = (layer == 0) ? 1 : 10;
  int Cout = (layer == 2) ? 1 : 10;
  int M = lane & 15, g = lane >> 4;
  int v = e >> 1, h = e & 1;
  int kloc = 16 * g + e; (void)v; (void)h;
  int gk  = kloc >> 4;
  int cin = kloc & 15;
  int tap = chunk + CHUNKS * gk;            // contiguous tap range per K-group
  float val = 0.f;
  if (tap < TAPS && cin < Cin && M < Cout) {
    int da = tap / 27; int rm = tap - da * 27;
    int db = rm / 9;   rm -= db * 9;
    int dc = rm / 3;   int dd = rm - dc * 3;
    int ta, tb, tc, td;
    if (perm) { ta = dc; tb = dd; tc = da; td = db; }   // w'[a,b,c,d] = w[c,d,a,b]
    else      { ta = da; tb = db; tc = dc; td = dd; }
    val = w[(size_t)(M * Cin + cin) * TAPS + ((ta * 3 + tb) * 3 + tc) * 3 + td];
  }
  wpack[tid] = (half_t)val;
}

// ---------------- misc: padded biases, tap byte-offset table, zero page ------
__global__ void pack_misc_kernel(const float* __restrict__ b1,
                                 const float* __restrict__ b2,
                                 const float* __restrict__ b3,
                                 float* __restrict__ biasPad,
                                 int* __restrict__ offTab,
                                 half_t* __restrict__ zpage) {
  int tid = threadIdx.x;
  if (tid < 48) {
    int layer = tid / 16, j = tid % 16;
    const float* b = (layer == 0) ? b1 : (layer == 1) ? b2 : b3;
    int Cout = (layer == 2) ? 1 : 10;
    biasPad[tid] = (j < Cout) ? b[j] : 0.f;
  } else if (tid < 48 + 2 * CHUNKS) {
    int t = tid - 48;
    int val = 0;
    if (t < TAPS) {
      int da = t / 27; int rm = t - da * 27;
      int db = rm / 9; rm -= db * 9;
      int dc = rm / 3; int dd = rm - dc * 3;
      val = ((da - 1) * 27000 + (db - 1) * 900 + (dc - 1) * 30 + (dd - 1)) * (CH * 2);
    }
    offTab[t] = val;
  } else if (tid < 48 + 2 * CHUNKS + 16) {
    zpage[tid - (48 + 2 * CHUNKS)] = (half_t)0.f;
  }
}

// ---------------- conv4d via implicit-GEMM WMMA ------------------------------
// MODE 0: write f16 activations [P][16] with bias+ReLU (layers 1,2)
// MODE 1: write/accumulate fp32 output channel 0 with bias+ReLU (layer 3)
template <int MODE>
__global__ void conv4d_wmma_kernel(const half_t* __restrict__ actIn,
                                   const half_t* __restrict__ wpack,
                                   const float* __restrict__ biasPad,
                                   const int* __restrict__ offTab,
                                   unsigned zoff,
                                   half_t* __restrict__ actOut,
                                   float* __restrict__ out,
                                   int accumulate) {
  int wave = blockIdx.x * (blockDim.x >> 5) + (threadIdx.x >> 5);
  if (wave >= NWAVES) return;               // wave-uniform
  int lane = threadIdx.x & 31;
  int n = lane & 15, g = lane >> 4;
  int tile0 = wave * JTILES;

  // ---- per-tile setup: 82-bit tap-validity mask (Kronecker of 4 dim masks) --
  unsigned long long mym[JTILES];
  unsigned boff[JTILES];
#pragma unroll
  for (int j = 0; j < JTILES; ++j) {
    int tile = tile0 + j;
    tile = tile < NTILES ? tile : NTILES - 1;   // clamp: duplicate compute OK
    int p = tile * 16 + n;
    int r = p;
    int bb = r / 810000;  r -= bb * 810000; (void)bb;
    int h1 = r / 27000;   r -= h1 * 27000;
    int w1 = r / 900;     r -= w1 * 900;
    int h2 = r / 30;
    int w2 = r - h2 * 30;
    unsigned mh1 = 2u | (unsigned)(h1 > 0) | ((h1 < SDIM - 1) ? 4u : 0u);
    unsigned mw1 = 2u | (unsigned)(w1 > 0) | ((w1 < SDIM - 1) ? 4u : 0u);
    unsigned mh2 = 2u | (unsigned)(h2 > 0) | ((h2 < SDIM - 1) ? 4u : 0u);
    unsigned mw2 = 2u | (unsigned)(w2 > 0) | ((w2 < SDIM - 1) ? 4u : 0u);
    unsigned m9 = ((mh2 & 1u) ? mw2 : 0u) | ((mh2 & 2u) ? (mw2 << 3) : 0u) |
                  ((mh2 & 4u) ? (mw2 << 6) : 0u);
    unsigned long long m27 =
        ((mw1 & 1u) ? (unsigned long long)m9 : 0ull) |
        ((mw1 & 2u) ? ((unsigned long long)m9 << 9) : 0ull) |
        ((mw1 & 4u) ? ((unsigned long long)m9 << 18) : 0ull);
    unsigned long long lo = 0ull; unsigned hi = 0u;
    if (mh1 & 1u) lo |= m27;
    if (mh1 & 2u) lo |= m27 << 27;
    if (mh1 & 4u) { lo |= m27 << 54; hi |= (unsigned)(m27 >> 10); }
    // lane group g handles taps [41g, 41g+41): shift its window to bit 0
    mym[j] = g ? ((lo >> 41) | ((unsigned long long)hi << 23)) : lo;
    boff[j] = (unsigned)p * 32u;            // byte offset of this position
  }

  const char* actC = (const char*)actIn;
  const char* wpC  = (const char*)wpack + (unsigned)lane * 32u;

  v8f acc[JTILES] = {};
#pragma unroll 1
  for (int c = 0; c < CHUNKS; ++c) {
    v16h a = *(const v16h*)(wpC + (unsigned)c * 1024u);
    int offv = g ? offTab[c + CHUNKS] : offTab[c];   // scalar loads + cndmask
    // phase 1: issue all B-fragment loads (distinct registers -> one clause)
    v16h bfv[JTILES];
#pragma unroll
    for (int j = 0; j < JTILES; ++j) {
      unsigned ok = (unsigned)(mym[j] >> c) & 1u;
      unsigned aoff = ok ? (boff[j] + (unsigned)offv) : zoff;
      bfv[j] = *(const v16h*)(actC + aoff);          // 32B, zero page if OOB
    }
    // phase 2: back-to-back WMMAs consume the pipelined loads
#pragma unroll
    for (int j = 0; j < JTILES; ++j) {
      acc[j] = __builtin_amdgcn_wmma_f32_16x16x32_f16(
          false, a, false, bfv[j], (short)0, acc[j], false, false);
    }
  }

  // ---- epilogue -------------------------------------------------------------
#pragma unroll
  for (int j = 0; j < JTILES; ++j) {
    int tile = tile0 + j;
    int tilec = tile < NTILES ? tile : NTILES - 1;
    int p = tilec * 16 + n;
    if (MODE == 0) {
      // clamped duplicates store identical data to the same address: benign
      v8h hv;
#pragma unroll
      for (int e = 0; e < 8; ++e) {
        float v = acc[j][e] + biasPad[8 * g + e];    // acc elem e = channel 8g+e
        v = v > 0.f ? v : 0.f;
        hv[e] = (half_t)v;
      }
      *(v8h*)((char*)actOut + (size_t)p * 32u + (unsigned)g * 16u) = hv;
    } else {
      if (g == 0 && tile < NTILES) {                 // guard: no double-accumulate
        float v = acc[j][0] + biasPad[0];            // channel 0 only (Cout=1)
        v = v > 0.f ? v : 0.f;
        if (accumulate) out[p] += v; else out[p] = v;
      }
    }
  }
}

// ---------------------------------------------------------------------------
extern "C" void kernel_launch(void* const* d_in, const int* in_sizes, int n_in,
                              void* d_out, int out_size, void* d_ws, size_t ws_size,
                              hipStream_t stream) {
  (void)in_sizes; (void)n_in; (void)out_size; (void)ws_size;
  const float* x  = (const float*)d_in[0];
  const float* w1 = (const float*)d_in[1];
  const float* b1 = (const float*)d_in[2];
  const float* w2 = (const float*)d_in[3];
  const float* b2 = (const float*)d_in[4];
  const float* w3 = (const float*)d_in[5];
  const float* b3 = (const float*)d_in[6];
  float* out = (float*)d_out;

  char* ws = (char*)d_ws;
  half_t* wpack   = (half_t*)(ws + WS_WPACK);
  float*  biasPad = (float*)(ws + WS_BIAS);
  int*    offTab  = (int*)(ws + WS_OFFTAB);
  half_t* act0    = (half_t*)(ws + WS_ACT0);
  half_t* act1    = (half_t*)(ws + WS_ACT1);
  half_t* act2    = (half_t*)(ws + WS_ACT2);
  half_t* zpage   = (half_t*)(ws + WS_ZPAGE);

  pack_weights_kernel<<<(6 * WBUF_HALVES + 255) / 256, 256, 0, stream>>>(w1, w2, w3, wpack);
  pack_misc_kernel<<<1, 256, 0, stream>>>(b1, b2, b3, biasPad, offTab, zpage);
  pack_input_kernel<<<(PTOT + 255) / 256, 256, 0, stream>>>(x, act0);

  const unsigned zoff0 = WS_ZPAGE - WS_ACT0;   // zero-page offset rel. to each act
  const unsigned zoff1 = WS_ZPAGE - WS_ACT1;
  const unsigned zoff2 = WS_ZPAGE - WS_ACT2;

  const int WAVES_PER_BLOCK = 4;
  const int blocks = (NWAVES + WAVES_PER_BLOCK - 1) / WAVES_PER_BLOCK;
  const int tpb = 32 * WAVES_PER_BLOCK;
  for (int br = 0; br < 2; ++br) {
    const half_t* wp = wpack + (size_t)br * 3 * WBUF_HALVES;
    conv4d_wmma_kernel<0><<<blocks, tpb, 0, stream>>>(
        act0, wp + 0 * WBUF_HALVES, biasPad + 0,  offTab, zoff0, act1, nullptr, 0);
    conv4d_wmma_kernel<0><<<blocks, tpb, 0, stream>>>(
        act1, wp + 1 * WBUF_HALVES, biasPad + 16, offTab, zoff1, act2, nullptr, 0);
    conv4d_wmma_kernel<1><<<blocks, tpb, 0, stream>>>(
        act2, wp + 2 * WBUF_HALVES, biasPad + 32, offTab, zoff2, nullptr, out, br);
  }
}
